// DeformableConv2d_15195594293677
// MI455X (gfx1250) — matrix operand order repaired and checked by reading, not compiled, not run on hardware
//
#include <hip/hip_runtime.h>
#include <hip/hip_bf16.h>

typedef __attribute__((ext_vector_type(16))) _Float16 v16h;
typedef __attribute__((ext_vector_type(8)))  _Float16 v8h;
typedef __attribute__((ext_vector_type(8)))  float    v8f;

namespace {
constexpr int BATCH = 4;
constexpr int CIN   = 64;
constexpr int COUT  = 128;
constexpr int Hd    = 128;
constexpr int Wd    = 128;
constexpr int HW    = Hd * Wd;        // 16384
constexpr int K2    = 9;
constexpr int KDIM  = CIN * K2;       // 576
constexpr int NPIX  = 32;             // pixels per block (N tile)
constexpr int NCHUNK = KDIM / 32;     // 18 k-chunks of 32
constexpr int BSTRIDE = 40;           // B-tile row stride in halves (80B: 16B-aligned, conflict-spread)
}

// ---------------------------------------------------------------------------
// Kernel 1: repack w_reg [Cout][Cin][3][3] f32 -> Wh [Cout][k] f16,
//           k = tap*64 + c, so each WMMA k-chunk is 32 contiguous halves/row.
// ---------------------------------------------------------------------------
__global__ __launch_bounds__(256) void repack_w(const float* __restrict__ w_reg,
                                                _Float16* __restrict__ Wh) {
    int i = blockIdx.x * 256 + threadIdx.x;
    if (i >= COUT * KDIM) return;
    int m = i / KDIM;
    int k = i - m * KDIM;
    int t = k >> 6;          // tap 0..8
    int c = k & 63;          // channel 0..63
    Wh[i] = (_Float16)w_reg[(m * CIN + c) * K2 + t];
}

// ---------------------------------------------------------------------------
// Kernel 2: fused offsets (18ch) + modulator (9ch) 3x3 conv, weights in LDS.
// One thread per output pixel, 27 register accumulators.
// ---------------------------------------------------------------------------
__global__ __launch_bounds__(256) void off_mod_conv(
    const float* __restrict__ x,     const float* __restrict__ w_off,
    const float* __restrict__ b_off, const float* __restrict__ w_mod,
    const float* __restrict__ b_mod, float* __restrict__ offBuf,
    float* __restrict__ modBuf) {
    __shared__ float wall[CIN * K2 * 27];   // [(c*9+tap)*27 + ch], 62 KB
    const int tid = threadIdx.x;
    for (int i = tid; i < CIN * K2 * 27; i += 256) {
        int ct = i / 27;          // c*9 + tap
        int ch = i - ct * 27;
        wall[i] = (ch < 18) ? w_off[ch * KDIM + ct] : w_mod[(ch - 18) * KDIM + ct];
    }
    __syncthreads();

    const int q   = blockIdx.x * 256 + tid;
    const int b   = q >> 14;
    const int rem = q & (HW - 1);
    const int y   = rem >> 7;
    const int xx  = rem & (Wd - 1);

    float acc[27];
#pragma unroll
    for (int ch = 0; ch < 27; ++ch) acc[ch] = 0.0f;

    const float* xb = x + (size_t)b * CIN * HW;
    for (int c = 0; c < CIN; ++c) {
        const float* xc = xb + c * HW;
#pragma unroll
        for (int tap = 0; tap < 9; ++tap) {
            const int yy = y + tap / 3 - 1;
            const int xq = xx + tap % 3 - 1;
            float v = 0.0f;
            if (yy >= 0 && yy < Hd && xq >= 0 && xq < Wd) v = xc[yy * Wd + xq];
            const int wb = (c * 9 + tap) * 27;
#pragma unroll
            for (int ch = 0; ch < 27; ++ch) acc[ch] += v * wall[wb + ch];
        }
    }
#pragma unroll
    for (int ch = 0; ch < 18; ++ch)
        offBuf[((b * 18 + ch) * Hd + y) * Wd + xx] = acc[ch] + b_off[ch];
#pragma unroll
    for (int ch = 0; ch < 9; ++ch) {
        float z = acc[18 + ch] + b_mod[ch];
        modBuf[((b * 9 + ch) * Hd + y) * Wd + xx] = 2.0f / (1.0f + __expf(-z));
    }
}

// ---------------------------------------------------------------------------
// Kernel 3: deformable conv as implicit GEMM with v_wmma_f32_16x16x32_f16.
// Block: 256 threads (8 wave32), tile = 128 Cout x 32 pixels.
// Double-buffered B-tile: one barrier per k-chunk; gather of chunk k+1 and
// A-fragment prefetch overlap the WMMAs of chunk k.
// ---------------------------------------------------------------------------
__global__ __launch_bounds__(256) void deform_wmma(
    const float* __restrict__ x, const float* __restrict__ offBuf,
    const float* __restrict__ modBuf, const _Float16* __restrict__ Wh,
    float* __restrict__ out) {
    __shared__ __align__(16) _Float16 Bs[2][NPIX * BSTRIDE]; // 2 x (32x32 f16, padded)
    __shared__ int   pL00[K2 * NPIX], pL01[K2 * NPIX];       // linear corner offsets
    __shared__ int   pL10[K2 * NPIX], pL11[K2 * NPIX];
    __shared__ float pW00[K2 * NPIX], pW01[K2 * NPIX];       // mask/mod-folded weights
    __shared__ float pW10[K2 * NPIX], pW11[K2 * NPIX];

    const int tid     = threadIdx.x;
    const int lane    = tid & 31;
    const int wave    = tid >> 5;          // 0..7 = Cout m-tile
    const int pixBase = blockIdx.x * NPIX;

    // ---- Phase 0: per-(tap,pixel) bilinear descriptors (once per block) ----
    for (int e = tid; e < K2 * NPIX; e += 256) {
        const int t  = e >> 5;             // tap
        const int p  = e & 31;             // pixel within tile
        const int q  = pixBase + p;
        const int b  = q >> 14;
        const int rm = q & (HW - 1);
        const int y  = rm >> 7;
        const int xx = rm & (Wd - 1);

        const float dy = offBuf[((b * 18 + 2 * t    ) * Hd + y) * Wd + xx];
        const float dx = offBuf[((b * 18 + 2 * t + 1) * Hd + y) * Wd + xx];
        const float mv = modBuf[((b * 9 + t) * Hd + y) * Wd + xx];

        const float py = (float)(y  + t / 3 - 1) + dy;
        const float px = (float)(xx + t % 3 - 1) + dx;
        const float fy0 = floorf(py), fx0 = floorf(px);
        const float ty = py - fy0, tx = px - fx0;
        const int y0 = (int)fy0, x0 = (int)fx0;

        const bool vy0 = (y0     >= 0) && (y0     <= Hd - 1);
        const bool vy1 = (y0 + 1 >= 0) && (y0 + 1 <= Hd - 1);
        const bool vx0 = (x0     >= 0) && (x0     <= Wd - 1);
        const bool vx1 = (x0 + 1 >= 0) && (x0 + 1 <= Wd - 1);

        const int cy0 = min(max(y0, 0), Hd - 1);
        const int cy1 = min(max(y0 + 1, 0), Hd - 1);
        const int cx0 = min(max(x0, 0), Wd - 1);
        const int cx1 = min(max(x0 + 1, 0), Wd - 1);

        pL00[e] = cy0 * Wd + cx0;
        pL01[e] = cy0 * Wd + cx1;
        pL10[e] = cy1 * Wd + cx0;
        pL11[e] = cy1 * Wd + cx1;
        pW00[e] = (1.0f - ty) * (1.0f - tx) * mv * ((vy0 && vx0) ? 1.0f : 0.0f);
        pW01[e] = (1.0f - ty) * tx          * mv * ((vy0 && vx1) ? 1.0f : 0.0f);
        pW10[e] = ty * (1.0f - tx)          * mv * ((vy1 && vx0) ? 1.0f : 0.0f);
        pW11[e] = ty * tx                   * mv * ((vy1 && vx1) ? 1.0f : 0.0f);
    }

    // Gather+interpolate one 32(k) x 32(px) chunk into a B-tile buffer.
    // Each thread produces two adjacent channels per descriptor fetch and
    // stores them as one packed 32-bit LDS write.
    auto build = [&](int ck, _Float16* B) {
        const int t  = ck >> 1;
        const int c0 = (ck & 1) << 5;
        for (int e = tid; e < 16 * NPIX; e += 256) {
            const int cl = e >> 5;         // channel-pair index 0..15
            const int p  = e & 31;
            const int pe = t * NPIX + p;
            const int q  = pixBase + p;
            const int b  = q >> 14;
            const float* xc0 = x + ((size_t)(b * CIN + c0 + 2 * cl)) * HW;
            const float* xc1 = xc0 + HW;
            const int   l00 = pL00[pe], l01 = pL01[pe];
            const int   l10 = pL10[pe], l11 = pL11[pe];
            const float w00 = pW00[pe], w01 = pW01[pe];
            const float w10 = pW10[pe], w11 = pW11[pe];
            const float s0 = w00 * xc0[l00] + w01 * xc0[l01]
                           + w10 * xc0[l10] + w11 * xc0[l11];
            const float s1 = w00 * xc1[l00] + w01 * xc1[l01]
                           + w10 * xc1[l10] + w11 * xc1[l11];
            union { _Float16 h[2]; unsigned u; } pk;
            pk.h[0] = (_Float16)s0;
            pk.h[1] = (_Float16)s1;
            *(unsigned*)(B + p * BSTRIDE + 2 * cl) = pk.u;
        }
    };

    const int half = lane >> 4;            // 0: low half-wave, 1: high
    const int nIn  = lane & 15;
    const _Float16* wrow = Wh + (size_t)(wave * 16 + nIn) * KDIM;

    union AF { v16h v; v8h h[2]; };
    // A (16x32) layout: elems 0..7 -> K = 8*half..+7 ; elems 8..15 -> K = 16+8*half..+7
    AF a_cur, a_nxt;
    a_cur.h[0] = *(const v8h*)(wrow +      8 * half);
    a_cur.h[1] = *(const v8h*)(wrow + 16 + 8 * half);
    a_nxt = a_cur;

    build(0, Bs[0]);           // also covered by the barrier below (params ready)
    __syncthreads();

    v8f acc0 = {};
    v8f acc1 = {};

    for (int ck = 0; ck < NCHUNK; ++ck) {
        const int cur = ck & 1;

        // B (32x16) layout: lane col = nIn, elems 0..7 -> K = 8*half..+7 ; elems 8..15 -> K = 16+8*half..+7
        const _Float16* bp0 = Bs[cur] + nIn * BSTRIDE + 8 * half;
        const _Float16* bp1 = Bs[cur] + (16 + nIn) * BSTRIDE + 8 * half;
        AF fb0, fb1;
        fb0.h[0] = *(const v8h*)(bp0);  fb0.h[1] = *(const v8h*)(bp0 + 16);
        fb1.h[0] = *(const v8h*)(bp1);  fb1.h[1] = *(const v8h*)(bp1 + 16);

        if (ck + 1 < NCHUNK) {
            // prefetch next A-fragment and overlap next chunk's gather
            const int kB = ((ck + 1) >> 1) * 64 + (((ck + 1) & 1) << 5);
            a_nxt.h[0] = *(const v8h*)(wrow + kB +      8 * half);
            a_nxt.h[1] = *(const v8h*)(wrow + kB + 16 + 8 * half);
            build(ck + 1, Bs[cur ^ 1]);
        }

        acc0 = __builtin_amdgcn_wmma_f32_16x16x32_f16(
            false, a_cur.v, false, fb0.v, (short)0, acc0, false, false);
        acc1 = __builtin_amdgcn_wmma_f32_16x16x32_f16(
            false, a_cur.v, false, fb1.v, (short)0, acc1, false, false);

        a_cur = a_nxt;
        __syncthreads();
    }

    // ---- epilogue: D layout -> lane col N = nIn (pixel), VGPR r -> M = r + 8*half ----
#pragma unroll
    for (int r = 0; r < 8; ++r) {
        const int m = wave * 16 + r + 8 * half;
        const int q0  = pixBase + nIn;
        const int b0i = q0 >> 14, rm0 = q0 & (HW - 1);
        out[((size_t)(b0i * COUT + m)) * HW + rm0] = acc0[r];
        const int q1  = q0 + 16;
        const int b1i = q1 >> 14, rm1 = q1 & (HW - 1);
        out[((size_t)(b1i * COUT + m)) * HW + rm1] = acc1[r];
    }
}

// ---------------------------------------------------------------------------
extern "C" void kernel_launch(void* const* d_in, const int* in_sizes, int n_in,
                              void* d_out, int out_size, void* d_ws, size_t ws_size,
                              hipStream_t stream) {
    (void)in_sizes; (void)n_in; (void)out_size; (void)ws_size;
    const float* x     = (const float*)d_in[0];
    const float* w_off = (const float*)d_in[1];
    const float* b_off = (const float*)d_in[2];
    const float* w_mod = (const float*)d_in[3];
    const float* b_mod = (const float*)d_in[4];
    const float* w_reg = (const float*)d_in[5];
    float* out = (float*)d_out;

    // Workspace layout: offsets f32 | modulators f32 | f16 repacked weights
    float*    offBuf = (float*)d_ws;                                 // 4*18*16384 floats
    float*    modBuf = offBuf + (size_t)BATCH * 18 * HW;             // 4*9*16384 floats
    _Float16* Wh     = (_Float16*)(modBuf + (size_t)BATCH * 9 * HW); // 128*576 halves

    repack_w<<<(COUT * KDIM + 255) / 256, 256, 0, stream>>>(w_reg, Wh);
    off_mod_conv<<<(BATCH * HW) / 256, 256, 0, stream>>>(x, w_off, b_off, w_mod,
                                                         b_mod, offBuf, modBuf);
    deform_wmma<<<(BATCH * HW) / NPIX, 256, 0, stream>>>(x, offBuf, modBuf, Wh, out);
}
